// LDC_68582037782958
// MI455X (gfx1250) — compile-verified, matrix operand order checked by reading, not run on hardware
//
#include <hip/hip_runtime.h>

// ---------------- types for WMMA ----------------
typedef __attribute__((ext_vector_type(16))) __bf16 v16bf;
typedef __attribute__((ext_vector_type(8)))  __bf16 v8bf;
typedef __attribute__((ext_vector_type(8)))  float  v8f;
typedef __attribute__((ext_vector_type(4)))  int    v4i;

__device__ __forceinline__ unsigned short f2bf(float f) {
    union { float f; unsigned u; } x; x.f = f;
    unsigned r = x.u + 0x7FFFu + ((x.u >> 16) & 1u);   // round-to-nearest-even
    return (unsigned short)(r >> 16);
}

__device__ __forceinline__ v16bf ld_bf16x16(const unsigned short* p) {
    v8bf lo = *(const v8bf*)(p);
    v8bf hi = *(const v8bf*)(p + 16);
    return __builtin_shufflevector(lo, hi, 0,1,2,3,4,5,6,7,8,9,10,11,12,13,14,15);
}

#define WMMA_BF16(A, B, C) \
    __builtin_amdgcn_wmma_f32_16x16x32_bf16(false, (A), false, (B), (short)0, (C), false, false)

// ---------------- async global->LDS staging (CDNA5 path if available) ----------------
#if __has_builtin(__builtin_amdgcn_global_load_async_to_lds_b128)
#define HAVE_ASYNC_LDS 1
__device__ __forceinline__ void g2l_b128(const unsigned short* gp, unsigned short* lp) {
    __builtin_amdgcn_global_load_async_to_lds_b128(
        (__attribute__((address_space(1))) v4i*)(uintptr_t)gp,
        (__attribute__((address_space(3))) v4i*)(unsigned)(uintptr_t)lp,
        0, 0);
}
__device__ __forceinline__ void wait_async() {
#if __has_builtin(__builtin_amdgcn_s_wait_asynccnt)
    __builtin_amdgcn_s_wait_asynccnt(0);
#else
    asm volatile("s_wait_asynccnt 0x0" ::: "memory");
#endif
}
#else
#define HAVE_ASYNC_LDS 0
__device__ __forceinline__ void g2l_b128(const unsigned short* gp, unsigned short* lp) {
    *(uint4*)lp = *(const uint4*)gp;
}
__device__ __forceinline__ void wait_async() {}
#endif

// ---------------- problem constants ----------------
#define BATCH 16
#define HW    16384      // 128*128

// workspace byte offsets (all 256-aligned)
#define WS_POOLED   0          // 16*256 f32            = 16384
#define WS_HK       16384      // 16*64  f32            = 4096
#define WS_HB       20480      // 16*32  f32            = 2048
#define WS_DYNB     22528      // 16*128 f32            = 8192
#define WS_INV      30720      // 128 f32               = 512
#define WS_B2       31232      // 128 f32               = 512
#define WS_ENW_BF   31744      // 128*256 bf16          = 65536
#define WS_DYNK_BF  97280      // 16*9*128*128 bf16     = 4718592
#define WS_ENH_BF   4815872    // 16*16384*128 bf16     = 67108864
// total ~71.9 MB

// ================= 1) global average pool =================
__global__ __launch_bounds__(256) void ldc_pool(const float* __restrict__ ir,
                                                const float* __restrict__ vi,
                                                float* __restrict__ pooled) {
    int bc = blockIdx.x;                  // 0..4095 : b*256 + c2
    int b = bc >> 8, c2 = bc & 255;
    const float* src = (c2 < 128)
        ? ir + (((size_t)(b * 128 + c2)) << 14)
        : vi + (((size_t)(b * 128 + (c2 - 128))) << 14);
    int tid = threadIdx.x;
    float s = 0.f;
    #pragma unroll
    for (int i = 0; i < 16; ++i) {
        float4 v = *(const float4*)(src + i * 1024 + tid * 4);
        s += v.x + v.y + v.z + v.w;
    }
    __shared__ float R[256];
    R[tid] = s; __syncthreads();
    for (int st = 128; st > 0; st >>= 1) {
        if (tid < st) R[tid] += R[tid + st];
        __syncthreads();
    }
    if (tid == 0) pooled[bc] = R[0] * (1.0f / 16384.0f);
}

// ================= 2) BN constant prep =================
__global__ __launch_bounds__(128) void ldc_bnprep(const float* g, const float* be,
                                                  const float* mu, const float* va,
                                                  float* inv, float* b2) {
    int t = threadIdx.x;
    float iv = g[t] * rsqrtf(va[t] + 1e-5f);
    inv[t] = iv;
    b2[t]  = be[t] - mu[t] * iv;
}

// ================= 3) convert en_w to bf16 =================
__global__ __launch_bounds__(256) void ldc_cvt_enw(const float* __restrict__ w,
                                                   unsigned short* __restrict__ wb) {
    int i = blockIdx.x * 256 + threadIdx.x;   // 128 blocks -> 32768 exact
    wb[i] = f2bf(w[i]);
}

// ================= 4) small MLP hidden layers =================
__global__ __launch_bounds__(128) void ldc_mlp(const float* __restrict__ pooled,
                                               const float* __restrict__ kg_w1, const float* __restrict__ kg_b1,
                                               const float* __restrict__ bg_w1, const float* __restrict__ bg_b1,
                                               float* __restrict__ hk, float* __restrict__ hb) {
    int b = blockIdx.x, tid = threadIdx.x;
    __shared__ float P[256];
    P[tid]       = pooled[b * 256 + tid];
    P[tid + 128] = pooled[b * 256 + tid + 128];
    __syncthreads();
    if (tid < 64) {
        const float* row = kg_w1 + (size_t)tid * 256;
        float s = kg_b1[tid];
        for (int j = 0; j < 256; ++j) s += P[j] * row[j];
        hk[b * 64 + tid] = fmaxf(s, 0.f);
    } else if (tid < 96) {
        int t = tid - 64;
        const float* row = bg_w1 + (size_t)t * 256;
        float s = bg_b1[t];
        for (int j = 0; j < 256; ++j) s += P[j] * row[j];
        hb[b * 32 + t] = fmaxf(s, 0.f);
    }
}

// ================= 5) dynamic kernel gen -> bf16 [b][tap][o][c] =================
__global__ __launch_bounds__(256) void ldc_dynk(const float* __restrict__ hk,
                                                const float* __restrict__ kg_w2,
                                                const float* __restrict__ kg_b2,
                                                unsigned short* __restrict__ dynk_bf) {
    __shared__ float LH[1024];                // hk: 16 x 64
    int tid = threadIdx.x;
    for (int i = tid; i < 1024; i += 256) LH[i] = hk[i];
    __syncthreads();
    int n = blockIdx.x * 256 + tid;           // 576 blocks -> 147456 exact
    const float* wr = kg_w2 + (size_t)n * 64;
    float wv[64];
    #pragma unroll
    for (int i = 0; i < 16; ++i) {
        float4 q = *(const float4*)(wr + i * 4);
        wv[i * 4 + 0] = q.x; wv[i * 4 + 1] = q.y;
        wv[i * 4 + 2] = q.z; wv[i * 4 + 3] = q.w;
    }
    float bias = kg_b2[n];
    int tap = n % 9, oc = n / 9;
    int c = oc & 127, o = oc >> 7;
    for (int bb = 0; bb < 16; ++bb) {
        const float* hh = &LH[bb * 64];
        float s = bias;
        #pragma unroll
        for (int j = 0; j < 64; ++j) s += hh[j] * wv[j];
        dynk_bf[(((size_t)bb * 9 + tap) * 128 + o) * 128 + c] = f2bf(s);
    }
}

// ================= 6) dynamic bias gen =================
__global__ __launch_bounds__(128) void ldc_dynb(const float* __restrict__ hb,
                                                const float* __restrict__ bg_w2,
                                                const float* __restrict__ bg_b2,
                                                float* __restrict__ dynb) {
    int b = blockIdx.x, tid = threadIdx.x;
    __shared__ float H[32];
    if (tid < 32) H[tid] = hb[b * 32 + tid];
    __syncthreads();
    const float* row = bg_w2 + (size_t)tid * 32;
    float s = bg_b2[tid];
    #pragma unroll
    for (int j = 0; j < 32; ++j) s += H[j] * row[j];
    dynb[b * 128 + tid] = s;
}

// ================= 7) enhance: 1x1 conv GEMM + BN + ReLU -> NHWC bf16 =================
// grid: (256 pixel tiles of 64, 16 batches), 256 threads = 8 waves.
// Block computes ALL 128 out-channels x 64 px: each wave holds 4 acc tiles,
// so each staged B tile is read once and feeds 4 WMMAs.
__global__ __launch_bounds__(256) void ldc_enhance(const float* __restrict__ ir,
                                                   const float* __restrict__ vi,
                                                   const unsigned short* __restrict__ enw_bf,
                                                   const float* __restrict__ inv,
                                                   const float* __restrict__ b2,
                                                   unsigned short* __restrict__ enh_bf) {
    const int b  = blockIdx.y;
    const int p0 = blockIdx.x * 64;
    const int tid = threadIdx.x;
    const int lane = tid & 31, wv = tid >> 5;
    const int ob0  = (wv >> 2) * 16;            // wave's base out-channel (j adds j*32)
    const int pn   = (wv & 3) * 16;             // wave's local pixel base
    const int m    = lane & 15;
    const int koff = (lane >> 4) * 8;

    __shared__ __align__(16) unsigned short Bt[64 * 40];   // [pixel][c_local], stride 40 halves

    v8f acc[4] = {v8f{}, v8f{}, v8f{}, v8f{}};
    for (int c0 = 0; c0 < 256; c0 += 32) {
        __syncthreads();
        {   // stage 32 chan x 64 px, transposed, fp32 -> bf16
            int cl = tid >> 3;                 // 0..31 channel-local
            int pg = (tid & 7) * 8;            // pixel group
            int cc = c0 + cl;
            const float* src = (cc < 128)
                ? ir + (((size_t)(b * 128 + cc)) << 14)
                : vi + (((size_t)(b * 128 + (cc - 128))) << 14);
            float4 v0 = *(const float4*)(src + p0 + pg);
            float4 v1 = *(const float4*)(src + p0 + pg + 4);
            Bt[(pg + 0) * 40 + cl] = f2bf(v0.x);
            Bt[(pg + 1) * 40 + cl] = f2bf(v0.y);
            Bt[(pg + 2) * 40 + cl] = f2bf(v0.z);
            Bt[(pg + 3) * 40 + cl] = f2bf(v0.w);
            Bt[(pg + 4) * 40 + cl] = f2bf(v1.x);
            Bt[(pg + 5) * 40 + cl] = f2bf(v1.y);
            Bt[(pg + 6) * 40 + cl] = f2bf(v1.z);
            Bt[(pg + 7) * 40 + cl] = f2bf(v1.w);
        }
        __syncthreads();
        v16bf B = ld_bf16x16(&Bt[(size_t)(pn + m) * 40 + koff]);
        #pragma unroll
        for (int j = 0; j < 4; ++j) {
            v16bf A = ld_bf16x16(enw_bf + (size_t)(ob0 + j * 32 + m) * 256 + c0 + koff);
            acc[j] = WMMA_BF16(A, B, acc[j]);
        }
    }
    // BN + ReLU, pack bf16 pairs, store NHWC
    const int p = p0 + pn + m;                 // this lane's pixel
    #pragma unroll
    for (int j = 0; j < 4; ++j) {
        const int obase = ob0 + j * 32 + koff; // out-channel for acc[j][r] is obase+r
        unsigned short* outp = enh_bf + ((size_t)b * HW + p) * 128 + obase;
        #pragma unroll
        for (int r = 0; r < 8; r += 2) {
            float v0 = acc[j][r]     * inv[obase + r]     + b2[obase + r];
            float v1 = acc[j][r + 1] * inv[obase + r + 1] + b2[obase + r + 1];
            v0 = v0 > 0.f ? v0 : 0.f;
            v1 = v1 > 0.f ? v1 : 0.f;
            unsigned pk = (unsigned)f2bf(v0) | ((unsigned)f2bf(v1) << 16);
            *(unsigned*)(outp + r) = pk;
        }
    }
}

// ================= 8) per-sample dynamic 3x3 conv (implicit GEMM) =================
// grid: (2 wtiles, 128 rows, 16 batches), 256 threads = 8 waves.
// Block tile: 128 out-ch x 64 px of one row; LDS slab: 3 rows x 66 px x 128 ch,
// zero padded, staged via async global->LDS when available. Each wave holds
// 4 acc tiles -> 144 WMMAs per wave, every B tile read feeds 4 WMMAs.
__global__ __launch_bounds__(256) void ldc_dynconv(const unsigned short* __restrict__ enh_bf,
                                                   const unsigned short* __restrict__ dynk_bf,
                                                   const float* __restrict__ dynb,
                                                   float* __restrict__ out) {
    const int b  = blockIdx.z;
    const int h  = blockIdx.y;
    const int w0 = blockIdx.x * 64;
    const int tid = threadIdx.x;

    __shared__ __align__(16) unsigned short S[3 * 66 * 136];  // [pos][c], stride 136 halves

    // stage zero-padded 3x66x128 slab of bf16 activations (16B chunks)
    for (int idx = tid; idx < 3 * 66 * 16; idx += 256) {
        int chunk = idx & 15, pos = idx >> 4;
        int dr = pos / 66, x = pos - dr * 66;
        int gh = h + dr - 1, gx = w0 + x - 1;
        unsigned short* lp = &S[(size_t)pos * 136 + chunk * 8];
        if ((unsigned)gh < 128u && (unsigned)gx < 128u) {
            g2l_b128(enh_bf + ((size_t)b * HW + gh * 128 + gx) * 128 + chunk * 8, lp);
        } else {
            *(uint4*)lp = make_uint4(0u, 0u, 0u, 0u);
        }
    }
    wait_async();
    __syncthreads();

    const int lane = tid & 31, wv = tid >> 5;
    const int ob0  = (wv >> 2) * 16;
    const int wl   = (wv & 3) * 16;
    const int m    = lane & 15;
    const int koff = (lane >> 4) * 8;

    v8f acc[4] = {v8f{}, v8f{}, v8f{}, v8f{}};
    #pragma unroll
    for (int tap = 0; tap < 9; ++tap) {
        const int kh = tap / 3, kw = tap - kh * 3;
        const unsigned short* Ab =
            dynk_bf + (((size_t)b * 9 + tap) * 128 + ob0 + m) * 128;
        const unsigned short* Brow = &S[(size_t)(kh * 66 + wl + m + kw) * 136];
        #pragma unroll
        for (int c0 = 0; c0 < 128; c0 += 32) {
            v16bf B = ld_bf16x16(Brow + c0 + koff);
            #pragma unroll
            for (int j = 0; j < 4; ++j) {
                v16bf A = ld_bf16x16(Ab + (size_t)j * 32 * 128 + c0 + koff);
                acc[j] = WMMA_BF16(A, B, acc[j]);
            }
        }
    }

    // epilogue: + dynamic bias, store NCHW fp32
    const int w = w0 + wl + m;
    #pragma unroll
    for (int j = 0; j < 4; ++j) {
        #pragma unroll
        for (int r = 0; r < 8; ++r) {
            int o = ob0 + j * 32 + koff + r;
            out[(((size_t)b * 128 + o) << 14) + h * 128 + w] = acc[j][r] + dynb[b * 128 + o];
        }
    }
}

// ================= launcher =================
extern "C" void kernel_launch(void* const* d_in, const int* in_sizes, int n_in,
                              void* d_out, int out_size, void* d_ws, size_t ws_size,
                              hipStream_t stream) {
    const float* ir    = (const float*)d_in[0];
    const float* vi    = (const float*)d_in[1];
    const float* kg_w1 = (const float*)d_in[2];
    const float* kg_b1 = (const float*)d_in[3];
    const float* kg_w2 = (const float*)d_in[4];
    const float* kg_b2 = (const float*)d_in[5];
    const float* bg_w1 = (const float*)d_in[6];
    const float* bg_b1 = (const float*)d_in[7];
    const float* bg_w2 = (const float*)d_in[8];
    const float* bg_b2 = (const float*)d_in[9];
    const float* en_w  = (const float*)d_in[10];
    const float* bn_g  = (const float*)d_in[11];
    const float* bn_b  = (const float*)d_in[12];
    const float* bn_m  = (const float*)d_in[13];
    const float* bn_v  = (const float*)d_in[14];
    float* outp = (float*)d_out;

    char* ws = (char*)d_ws;
    float*          pooled  = (float*)(ws + WS_POOLED);
    float*          hk      = (float*)(ws + WS_HK);
    float*          hb      = (float*)(ws + WS_HB);
    float*          dynb    = (float*)(ws + WS_DYNB);
    float*          inv     = (float*)(ws + WS_INV);
    float*          b2      = (float*)(ws + WS_B2);
    unsigned short* enw_bf  = (unsigned short*)(ws + WS_ENW_BF);
    unsigned short* dynk_bf = (unsigned short*)(ws + WS_DYNK_BF);
    unsigned short* enh_bf  = (unsigned short*)(ws + WS_ENH_BF);

    ldc_pool   <<<dim3(4096), 256, 0, stream>>>(ir, vi, pooled);
    ldc_bnprep <<<dim3(1),    128, 0, stream>>>(bn_g, bn_b, bn_m, bn_v, inv, b2);
    ldc_cvt_enw<<<dim3(128),  256, 0, stream>>>(en_w, enw_bf);
    ldc_mlp    <<<dim3(16),   128, 0, stream>>>(pooled, kg_w1, kg_b1, bg_w1, bg_b1, hk, hb);
    ldc_dynk   <<<dim3(576),  256, 0, stream>>>(hk, kg_w2, kg_b2, dynk_bf);
    ldc_dynb   <<<dim3(16),   128, 0, stream>>>(hb, bg_w2, bg_b2, dynb);
    ldc_enhance<<<dim3(256, 16),    256, 0, stream>>>(ir, vi, enw_bf, inv, b2, enh_bf);
    ldc_dynconv<<<dim3(2, 128, 16), 256, 0, stream>>>(enh_bf, dynk_bf, dynb, outp);
}
